// CopyGenerator_66803921322635
// MI455X (gfx1250) — matrix operand order repaired and checked by reading, compiled, not run on hardware
//
#include <hip/hip_runtime.h>
#include <hip/hip_bf16.h>
#include <math.h>

// Problem constants (from reference)
#define BB 8
#define HH 16
#define TT 128
#define SS 512
#define DD 1024
#define VV 32000
#define BT (BB*TT)   // 1024 rows

typedef __attribute__((ext_vector_type(16))) __bf16 v16bf;
typedef __attribute__((ext_vector_type(8)))  __bf16 v8bf;
typedef __attribute__((ext_vector_type(8)))  float  v8f;

__device__ inline v8f wmma_bf16(v16bf a, v16bf b, v8f c) {
  return __builtin_amdgcn_wmma_f32_16x16x32_bf16(false, a, false, b, (short)0, c, false, false);
}

// ---------------- head mean: attn[B,H,T,S] -> attn_mean[B,T,S] ----------------
__global__ void head_mean_kernel(const float* __restrict__ attn, float* __restrict__ out) {
  int i = blockIdx.x * blockDim.x + threadIdx.x;       // over B*T*S
  if (i >= BB*TT*SS) return;
  int b  = i / (TT*SS);
  int ts = i - b*(TT*SS);
  float s = 0.f;
  #pragma unroll
  for (int h = 0; h < HH; ++h)
    s += attn[((size_t)b*HH + h)*(TT*SS) + ts];
  out[i] = s * (1.0f/16.0f);
}

// ---------------- f32 -> bf16 hi/lo split ----------------
__global__ void split_bf16x2_kernel(const float* __restrict__ x,
                                    __bf16* __restrict__ hi, __bf16* __restrict__ lo, int n) {
  int i = blockIdx.x * blockDim.x + threadIdx.x;
  if (i >= n) return;
  float v = x[i];
  __bf16 h = (__bf16)v;
  hi[i] = h;
  lo[i] = (__bf16)(v - (float)h);
}

// ---------------- per-row dot with a shared vector: out[r] = <X[r,:K], w> ----------------
__global__ void row_dot_kernel(const float* __restrict__ X, const float* __restrict__ w,
                               float* __restrict__ out, int rows, int K) {
  int row  = blockIdx.x * (blockDim.x >> 5) + (threadIdx.x >> 5);
  int lane = threadIdx.x & 31;
  if (row >= rows) return;
  float s = 0.f;
  for (int k = lane; k < K; k += 32) s += X[(size_t)row*K + k] * w[k];
  #pragma unroll
  for (int off = 16; off; off >>= 1) s += __shfl_down(s, off, 32);
  if (lane == 0) out[row] = s;
}

// ---------------- copy gate: prob[row] = sigmoid( sum_s attn*mw + dw + b ) ----------------
__global__ void gate_kernel(const float* __restrict__ attn_mean, const float* __restrict__ mw,
                            const float* __restrict__ dw, const float* __restrict__ bp,
                            float* __restrict__ prob) {
  int row  = blockIdx.x * (blockDim.x >> 5) + (threadIdx.x >> 5);  // B*T rows
  int lane = threadIdx.x & 31;
  if (row >= BT) return;
  int b = row / TT;
  float s = 0.f;
  for (int k = lane; k < SS; k += 32)
    s += attn_mean[(size_t)row*SS + k] * mw[b*SS + k];
  #pragma unroll
  for (int off = 16; off; off >>= 1) s += __shfl_down(s, off, 32);
  if (lane == 0) {
    float x = s + dw[row] + bp[0];
    prob[row] = 1.0f / (1.0f + expf(-x));
  }
}

// ---------------- bf16x3 WMMA GEMM: out[m,n] = sum_k A[m,k]*W[n,k] + bias[n] ----------------
// A: [BT, DD] hi/lo bf16 ; W: [VV, DD] hi/lo bf16 ; out: [BT, VV] f32
// Wave tile 32M x 32N (4 accumulators); workgroup = 8 waves as 2(M) x 4(N) -> 64M x 128N.
__global__ __launch_bounds__(256)
void gemm_bf16x3_kernel(const __bf16* __restrict__ Ahi, const __bf16* __restrict__ Alo,
                        const __bf16* __restrict__ Whi, const __bf16* __restrict__ Wlo,
                        const float* __restrict__ bias, float* __restrict__ out) {
  const int tid  = threadIdx.x;
  const int wave = tid >> 5;          // 0..7
  const int lane = tid & 31;
  const int wm   = wave >> 2;         // 0..1 along M
  const int wn   = wave & 3;          // 0..3 along N
  const int tileM = blockIdx.x & 15;          // 16 M-tiles of 64
  const int tileN = blockIdx.x >> 4;          // 250 N-tiles of 128
  const int m0 = tileM*64  + wm*32;
  const int n0 = tileN*128 + wn*32;

  const int lm    = lane & 15;
  const int lhalf = lane >> 4;        // 0/1

  const __bf16* ah[2]; const __bf16* al[2];
  const __bf16* bh[2]; const __bf16* bl[2];
  #pragma unroll
  for (int m = 0; m < 2; ++m) {
    ah[m] = Ahi + (size_t)(m0 + m*16 + lm)*DD;
    al[m] = Alo + (size_t)(m0 + m*16 + lm)*DD;
  }
  #pragma unroll
  for (int n = 0; n < 2; ++n) {
    bh[n] = Whi + (size_t)(n0 + n*16 + lm)*DD;
    bl[n] = Wlo + (size_t)(n0 + n*16 + lm)*DD;
  }

  v8f acc[2][2] = {{{}, {}}, {{}, {}}};

  for (int k0 = 0; k0 < DD; k0 += 32) {
    const int ka = k0 + lhalf*8;    // A: K = ka+[0..7] and ka+16+[0..7]
    const int kb = k0 + lhalf*8;    // B: K = kb+[0..7] and kb+16+[0..7]
    if (k0 + 32 < DD) {             // prefetch next B tiles (global_prefetch_b8)
      __builtin_prefetch((const void*)(bh[0] + kb + 32), 0, 1);
      __builtin_prefetch((const void*)(bh[1] + kb + 32), 0, 1);
    }
    v16bf fah[2], fal[2], fbh[2], fbl[2];
    #pragma unroll
    for (int m = 0; m < 2; ++m) {
      v8bf p0 = *(const v8bf*)(ah[m] + ka);
      v8bf p1 = *(const v8bf*)(ah[m] + ka + 16);
      v8bf q0 = *(const v8bf*)(al[m] + ka);
      v8bf q1 = *(const v8bf*)(al[m] + ka + 16);
      #pragma unroll
      for (int i = 0; i < 8; ++i) {
        fah[m][i]=p0[i]; fah[m][8+i]=p1[i];
        fal[m][i]=q0[i]; fal[m][8+i]=q1[i];
      }
    }
    #pragma unroll
    for (int n = 0; n < 2; ++n) {
      v8bf r0 = *(const v8bf*)(bh[n] + kb);
      v8bf r1 = *(const v8bf*)(bh[n] + kb + 16);
      v8bf s0 = *(const v8bf*)(bl[n] + kb);
      v8bf s1 = *(const v8bf*)(bl[n] + kb + 16);
      #pragma unroll
      for (int i = 0; i < 8; ++i) {
        fbh[n][i]=r0[i]; fbh[n][8+i]=r1[i];
        fbl[n][i]=s0[i]; fbl[n][8+i]=s1[i];
      }
    }
    // bf16x3: hi*hi + hi*lo + lo*hi ; 12 WMMAs reuse 8 fragments
    #pragma unroll
    for (int m = 0; m < 2; ++m) {
      #pragma unroll
      for (int n = 0; n < 2; ++n) {
        acc[m][n] = wmma_bf16(fah[m], fbh[n], acc[m][n]);
        acc[m][n] = wmma_bf16(fah[m], fbl[n], acc[m][n]);
        acc[m][n] = wmma_bf16(fal[m], fbh[n], acc[m][n]);
      }
    }
  }

  // C/D layout: element r -> row = mbase + r + lhalf*8, col = nbase + lm
  #pragma unroll
  for (int n = 0; n < 2; ++n) {
    const int col = n0 + n*16 + lm;
    const float bia = bias[col];
    #pragma unroll
    for (int m = 0; m < 2; ++m) {
      #pragma unroll
      for (int r = 0; r < 8; ++r) {
        int row = m0 + m*16 + r + lhalf*8;
        out[(size_t)row*VV + col] = acc[m][n][r] + bia;
      }
    }
  }
}

// ---------------- block reductions ----------------
__device__ inline float blockReduceSum(float v, float* red) {
  int lane = threadIdx.x & 31, wid = threadIdx.x >> 5;
  #pragma unroll
  for (int off = 16; off; off >>= 1) v += __shfl_down(v, off, 32);
  if (lane == 0) red[wid] = v;
  __syncthreads();
  int nw = blockDim.x >> 5;
  v = (threadIdx.x < (unsigned)nw) ? red[threadIdx.x] : 0.f;
  if (wid == 0) {
    #pragma unroll
    for (int off = 16; off; off >>= 1) v += __shfl_down(v, off, 32);
    if (lane == 0) red[0] = v;
  }
  __syncthreads();
  float r = red[0];
  __syncthreads();
  return r;
}

__device__ inline float blockReduceMax(float v, float* red) {
  int lane = threadIdx.x & 31, wid = threadIdx.x >> 5;
  #pragma unroll
  for (int off = 16; off; off >>= 1) v = fmaxf(v, __shfl_down(v, off, 32));
  if (lane == 0) red[wid] = v;
  __syncthreads();
  int nw = blockDim.x >> 5;
  v = (threadIdx.x < (unsigned)nw) ? red[threadIdx.x] : -3.4e38f;
  if (wid == 0) {
    #pragma unroll
    for (int off = 16; off; off >>= 1) v = fmaxf(v, __shfl_down(v, off, 32));
    if (lane == 0) red[0] = v;
  }
  __syncthreads();
  float r = red[0];
  __syncthreads();
  return r;
}

// ---------------- finalize: per-row dense LDS copy-scatter + fused double softmax + log ----------------
__global__ void finalize_kernel(const float* __restrict__ attn_mean, const int* __restrict__ src,
                                const float* __restrict__ prob, float* __restrict__ out) {
  extern __shared__ float copyv[];     // VV floats = 125 KB (CDNA5: up to 320 KB/WG)
  __shared__ float red[32];
  const int row = blockIdx.x;          // 0..BT-1
  const int b   = row / TT;
  const int tid = threadIdx.x;
  const int nt  = blockDim.x;

  for (int v = tid; v < VV; v += nt) copyv[v] = 0.f;
  __syncthreads();

  // scatter-add attention mass onto vocab ids (ds_add_f32 atomics)
  for (int s = tid; s < SS; s += nt)
    atomicAdd(&copyv[src[b*SS + s]], attn_mean[(size_t)row*SS + s]);
  __syncthreads();

  // copy-distribution softmax stats (zeros participate)
  float mc = 0.f;
  for (int v = tid; v < VV; v += nt) mc = fmaxf(mc, copyv[v]);
  mc = blockReduceMax(mc, red);
  float zc = 0.f;
  for (int v = tid; v < VV; v += nt) zc += expf(copyv[v] - mc);
  zc = blockReduceSum(zc, red);

  // generation logits softmax stats (row currently holds gen_logits)
  const float* g = out + (size_t)row*VV;
  float mg = -3.4e38f;
  for (int v = tid; v < VV; v += nt) mg = fmaxf(mg, g[v]);
  mg = blockReduceMax(mg, red);
  float zg = 0.f;
  for (int v = tid; v < VV; v += nt) zg += expf(g[v] - mg);
  zg = blockReduceSum(zg, red);

  const float p  = prob[row];
  const float sg = p / zg;
  const float sc = (1.0f - p) / zc;

  float* o = out + (size_t)row*VV;
  for (int v = tid; v < VV; v += nt) {
    float val = sg * expf(o[v] - mg) + sc * expf(copyv[v] - mc);
    o[v] = logf(val);
  }
}

// ---------------- launch ----------------
extern "C" void kernel_launch(void* const* d_in, const int* in_sizes, int n_in,
                              void* d_out, int out_size, void* d_ws, size_t ws_size,
                              hipStream_t stream) {
  (void)in_sizes; (void)n_in; (void)out_size; (void)ws_size;
  const int*   src  = (const int*)  d_in[0];   // [B,S]
  const float* dec  = (const float*)d_in[1];   // [B,T,D]
  const float* attn = (const float*)d_in[2];   // [B,H,T,S]
  const float* mem  = (const float*)d_in[3];   // [B,S,D]
  const float* Wg   = (const float*)d_in[4];   // [V,D]
  const float* bg   = (const float*)d_in[5];   // [V]
  const float* Wp   = (const float*)d_in[6];   // [1,2D]
  const float* bp   = (const float*)d_in[7];   // [1]
  float* out = (float*)d_out;                  // [B*T, V]

  // workspace carve (256B aligned)
  char* p = (char*)d_ws;
  auto carve = [&](size_t bytes) { char* q = p; p += (bytes + 255) & ~(size_t)255; return q; };
  float*  attn_mean = (float*) carve((size_t)BT*SS*4);        // 2 MB
  float*  mw        = (float*) carve((size_t)BB*SS*4);
  float*  dw        = (float*) carve((size_t)BT*4);
  float*  prob      = (float*) carve((size_t)BT*4);
  __bf16* a_hi      = (__bf16*)carve((size_t)BT*DD*2);        // 2 MB
  __bf16* a_lo      = (__bf16*)carve((size_t)BT*DD*2);
  __bf16* w_hi      = (__bf16*)carve((size_t)VV*DD*2);        // 65.5 MB (hi+lo stay L2-resident)
  __bf16* w_lo      = (__bf16*)carve((size_t)VV*DD*2);

  // 1) mean over heads
  head_mean_kernel<<<(BB*TT*SS + 255)/256, 256, 0, stream>>>(attn, attn_mean);
  // 2) bf16 hi/lo splits
  split_bf16x2_kernel<<<(BT*DD + 255)/256, 256, 0, stream>>>(dec, a_hi, a_lo, BT*DD);
  split_bf16x2_kernel<<<((int)((size_t)VV*DD) + 255)/256, 256, 0, stream>>>(Wg, w_hi, w_lo, VV*DD);
  // 3) gate pieces: mw[b,s] = <memory[b,s,:], Wp[:D]>, dw[row] = <dec[row,:], Wp[D:]>
  row_dot_kernel<<<(BB*SS)/8, 256, 0, stream>>>(mem, Wp,       mw, BB*SS, DD);
  row_dot_kernel<<<BT/8,      256, 0, stream>>>(dec, Wp + DD,  dw, BT,    DD);
  // 4) gate
  gate_kernel<<<BT/8, 256, 0, stream>>>(attn_mean, mw, dw, bp, prob);
  // 5) big GEMM (gen_logits + b_gen) -> d_out
  gemm_bf16x3_kernel<<<(BT/64) * (VV/128), 256, 0, stream>>>(a_hi, a_lo, w_hi, w_lo, bg, out);
  // 6) fused copy-scatter + double softmax + log, in place on d_out
  finalize_kernel<<<BT, 512, VV*sizeof(float), stream>>>(attn_mean, src, prob, out);
}
